// SelfAttention_3710851744401
// MI455X (gfx1250) — matrix operand order repaired and checked by reading, not compiled, not run on hardware
//
#include <hip/hip_runtime.h>
#include <hip/hip_bf16.h>

typedef unsigned short u16;
typedef unsigned int   u32;
typedef __attribute__((ext_vector_type(4)))  u32    u32x4;
typedef __attribute__((ext_vector_type(16))) __bf16 v16bf;
typedef __attribute__((ext_vector_type(8)))  float  v8f;

union Frag16 { v16bf bf; u32x4 u4[2]; };

#define NEG_BIG (-3.0e38f)

static __device__ inline u16 f32_to_bf16(float f) {
    u32 u = __float_as_uint(f);
    u32 r = 0x7FFFu + ((u >> 16) & 1u);
    return (u16)((u + r) >> 16);
}

static __device__ inline v8f wmma_bf16(const Frag16& a, const Frag16& b, v8f c) {
    return __builtin_amdgcn_wmma_f32_16x16x32_bf16(
        /*neg_a=*/false, a.bf, /*neg_b=*/false, b.bf,
        /*c_mod=*/(short)0, c, /*reuse_a=*/false, /*reuse_b=*/false);
}

// ---------------- f32 -> bf16 conversion ----------------
__global__ __launch_bounds__(256) void cvt_bf16_kernel(const float* __restrict__ src,
                                                       u16* __restrict__ dst, int n) {
    int i = blockIdx.x * 256 + threadIdx.x;
    if (i < n) dst[i] = f32_to_bf16(src[i]);
}

// ---------------- bf16 GEMM:  C[M,N] = A[M,K] * W[N,K]^T + bias ----------------
// block tile 128x64, BK=32, 256 threads = 8 waves, each wave: 16 rows x 4 WMMA n-tiles
template <bool OUT_F32>
__global__ __launch_bounds__(256)
void gemm_bf16_kernel(const u16* __restrict__ A, const u16* __restrict__ W,
                      const float* __restrict__ bias, void* __restrict__ Cout,
                      int M, int N, int K) {
    __shared__ __align__(16) u16 lsA[128 * 32];
    __shared__ __align__(16) u16 lsB[64 * 32];

    const int t    = threadIdx.x;
    const int wave = t >> 5;
    const int lane = t & 31;
    const int half = lane >> 4;
    const int l16  = lane & 15;

    const int blockM = blockIdx.y * 128;
    const int blockN = blockIdx.x * 64;

    v8f acc[4] = {};

    for (int kt = 0; kt < K; kt += 32) {
        { // A tile: 128x32, each thread 16 elems (2 x b128)
            const int row = t >> 1, seg = (t & 1) * 16;
            const u32x4* src = (const u32x4*)(A + (size_t)(blockM + row) * K + kt + seg);
            u32x4* dst = (u32x4*)(lsA + row * 32 + seg);
            dst[0] = src[0];
            dst[1] = src[1];
        }
        if (t < 128) { // W tile: 64x32
            const int row = t >> 1, seg = (t & 1) * 16;
            const u32x4* src = (const u32x4*)(W + (size_t)(blockN + row) * K + kt + seg);
            u32x4* dst = (u32x4*)(lsB + row * 32 + seg);
            dst[0] = src[0];
            dst[1] = src[1];
        }
        __syncthreads();

        Frag16 a;
        { // A fragment: row = l16 within wave strip; K runs at half*8 and 16+half*8
            const u16* p = lsA + (wave * 16 + l16) * 32 + half * 8;
            a.u4[0] = *(const u32x4*)p;
            a.u4[1] = *(const u32x4*)(p + 16);
        }
#pragma unroll
        for (int nt = 0; nt < 4; ++nt) {
            Frag16 b;
            const u16* p = lsB + (nt * 16 + l16) * 32 + half * 8;
            b.u4[0] = *(const u32x4*)p;
            b.u4[1] = *(const u32x4*)(p + 16);
            acc[nt] = wmma_bf16(a, b, acc[nt]);
        }
        __syncthreads();
    }

#pragma unroll
    for (int nt = 0; nt < 4; ++nt) {
        const int col = blockN + nt * 16 + l16;
        const float bv = bias ? bias[col] : 0.0f;
#pragma unroll
        for (int r = 0; r < 8; ++r) {
            const int row = blockM + wave * 16 + half * 8 + r;
            const float v = acc[nt][r] + bv;
            if (OUT_F32) ((float*)Cout)[(size_t)row * N + col] = v;
            else         ((u16*)Cout)[(size_t)row * N + col]   = f32_to_bf16(v);
        }
    }
}

// ---------------- flash attention ----------------
// qkv: bf16 [B*S, 3072] rows = {q(1024) | k(1024) | v(1024)}, per head 64 cols
// attn: bf16 [B*S, 1024] merged heads
__global__ __launch_bounds__(256)
void attn_kernel(const u16* __restrict__ qkv, u16* __restrict__ attn, int S) {
    const int D3 = 3072;
    const int bh = blockIdx.y;
    const int b = bh >> 4, h = bh & 15;
    const int qblk = blockIdx.x;

    const int t    = threadIdx.x;
    const int wave = t >> 5;
    const int lane = t & 31;
    const int half = lane >> 4;
    const int l16  = lane & 15;
    const int qrow0 = qblk * 128 + wave * 16;

    __shared__ __align__(16) u16 lsVt[64 * 32];     // V transposed: [dh][kv]
    __shared__ __align__(16) u16 lsP[8][16 * 32];   // per-wave P (C->A relayout)

    // Q fragments (registers, loaded once): 16 rows x dh64 = two 16x32 A-fragments
    Frag16 aQ[2];
    {
        const u16* qb = qkv + (size_t)(b * S + qrow0 + l16) * D3 + h * 64;
#pragma unroll
        for (int kp = 0; kp < 2; ++kp) {
            aQ[kp].u4[0] = *(const u32x4*)(qb + kp * 32 + half * 8);
            aQ[kp].u4[1] = *(const u32x4*)(qb + kp * 32 + 16 + half * 8);
        }
    }

    float m_s[8], l_s[8];
    v8f o[4] = {};
#pragma unroll
    for (int r = 0; r < 8; ++r) { m_s[r] = NEG_BIG; l_s[r] = 0.0f; }

    const int   kv_end = qblk * 128 + 128;  // causal upper bound for this q block
    const float scale  = 0.125f;            // 1/sqrt(64)

    for (int kt = 0; kt < kv_end; kt += 32) {
        { // cooperative V-tile load, transposed into LDS
            const int row = t >> 3;            // kv 0..31
            const int seg = (t & 7) * 8;       // dh segment
            const u16* vs = qkv + (size_t)(b * S + kt + row) * D3 + 2048 + h * 64 + seg;
            u32x4 d = *(const u32x4*)vs;
            const u16* e = (const u16*)&d;
#pragma unroll
            for (int j = 0; j < 8; ++j) lsVt[(seg + j) * 32 + row] = e[j];
        }
        __syncthreads();

        // scores: S[16x32] = Q(16x64) . K(32kv x 64)^T ; K B-fragments straight from global
        v8f s0 = {}, s1 = {};
        const u16* kb = qkv + (size_t)(b * S + kt) * D3 + 1024 + h * 64;
#pragma unroll
        for (int kp = 0; kp < 2; ++kp) {
            Frag16 bk0, bk1;
            const u16* p0 = kb + (size_t)l16 * D3 + kp * 32 + half * 8;
            bk0.u4[0] = *(const u32x4*)p0;
            bk0.u4[1] = *(const u32x4*)(p0 + 16);
            const u16* p1 = kb + (size_t)(16 + l16) * D3 + kp * 32 + half * 8;
            bk1.u4[0] = *(const u32x4*)p1;
            bk1.u4[1] = *(const u32x4*)(p1 + 16);
            s0 = wmma_bf16(aQ[kp], bk0, s0);
            s1 = wmma_bf16(aQ[kp], bk1, s1);
        }

        // online softmax (per q row; rows live in C-layout r+half*8, cols across 16 lanes)
        float p0v[8], p1v[8];
#pragma unroll
        for (int r = 0; r < 8; ++r) {
            const int qi = qrow0 + half * 8 + r;
            const int c0 = kt + l16, c1 = kt + 16 + l16;
            float x0 = (c0 <= qi) ? s0[r] * scale : NEG_BIG;
            float x1 = (c1 <= qi) ? s1[r] * scale : NEG_BIG;
            float mx = fmaxf(x0, x1);
#pragma unroll
            for (int off = 1; off < 16; off <<= 1) mx = fmaxf(mx, __shfl_xor(mx, off, 32));
            const float mnew  = fmaxf(m_s[r], mx);
            const float alpha = __expf(m_s[r] - mnew);
            const float e0 = __expf(x0 - mnew);
            const float e1 = __expf(x1 - mnew);
            float sum = e0 + e1;
#pragma unroll
            for (int off = 1; off < 16; off <<= 1) sum += __shfl_xor(sum, off, 32);
            l_s[r] = l_s[r] * alpha + sum;
            m_s[r] = mnew;
            p0v[r] = e0;
            p1v[r] = e1;
#pragma unroll
            for (int nt = 0; nt < 4; ++nt) o[nt][r] *= alpha;
        }

        // relayout P (C-layout) -> A-fragment via per-wave LDS buffer
        u16* myP = lsP[wave];
#pragma unroll
        for (int r = 0; r < 8; ++r) {
            myP[(half * 8 + r) * 32 + l16]      = f32_to_bf16(p0v[r]);
            myP[(half * 8 + r) * 32 + 16 + l16] = f32_to_bf16(p1v[r]);
        }
        Frag16 aP;
        {
            const u16* pp = myP + (size_t)l16 * 32 + half * 8;
            aP.u4[0] = *(const u32x4*)pp;
            aP.u4[1] = *(const u32x4*)(pp + 16);
        }
        // O(16x64) += P(16x32) . V(32x64); V B-fragments from transposed LDS
#pragma unroll
        for (int nt = 0; nt < 4; ++nt) {
            Frag16 bv;
            const u16* vp = lsVt + (nt * 16 + l16) * 32 + half * 8;
            bv.u4[0] = *(const u32x4*)vp;
            bv.u4[1] = *(const u32x4*)(vp + 16);
            o[nt] = wmma_bf16(aP, bv, o[nt]);
        }
        __syncthreads();  // protect lsVt before next tile overwrite
    }

    // epilogue: normalize and store merged-head bf16
#pragma unroll
    for (int nt = 0; nt < 4; ++nt) {
#pragma unroll
        for (int r = 0; r < 8; ++r) {
            const int row = qrow0 + half * 8 + r;
            const float v = o[nt][r] / l_s[r];
            attn[(size_t)(b * S + row) * 1024 + h * 64 + nt * 16 + l16] = f32_to_bf16(v);
        }
    }
}

// ---------------- launch ----------------
extern "C" void kernel_launch(void* const* d_in, const int* in_sizes, int n_in,
                              void* d_out, int out_size, void* d_ws, size_t ws_size,
                              hipStream_t stream) {
    const float* x     = (const float*)d_in[0];
    const float* w_in  = (const float*)d_in[1];
    const float* b_in  = (const float*)d_in[2];
    const float* w_out = (const float*)d_in[3];
    const float* b_out = (const float*)d_in[4];
    float* out = (float*)d_out;

    const int B = 4, S = 2048, D = 1024;
    const int M = B * S;  // 8192

    char* ws = (char*)d_ws;
    u16* xb   = (u16*)(ws + 0);                     // 16 MB  : x bf16 [8192,1024]
    u16* wib  = (u16*)(ws + (size_t)16777216);      //  6 MB  : w_in bf16 [3072,1024]
    u16* wob  = (u16*)(ws + (size_t)23068672);      //  2 MB  : w_out bf16 [1024,1024]
    u16* qkvb = (u16*)(ws + (size_t)25165824);      // 48 MB  : qkv bf16 [8192,3072]
    u16* attb = (u16*)(ws + (size_t)75497472);      // 16 MB  : attn bf16 [8192,1024]

    cvt_bf16_kernel<<<(M * D + 255) / 256, 256, 0, stream>>>(x, xb, M * D);
    cvt_bf16_kernel<<<(3 * D * D + 255) / 256, 256, 0, stream>>>(w_in, wib, 3 * D * D);
    cvt_bf16_kernel<<<(D * D + 255) / 256, 256, 0, stream>>>(w_out, wob, D * D);

    dim3 g1(3 * D / 64, M / 128);  // (48, 64)
    gemm_bf16_kernel<false><<<g1, 256, 0, stream>>>(xb, wib, b_in, qkvb, M, 3 * D, D);

    dim3 g2(S / 128, B * 16);      // (16, 64): q-blocks x (B*H)
    attn_kernel<<<g2, 256, 0, stream>>>(qkvb, attb, S);

    dim3 g3(D / 64, M / 128);      // (16, 64)
    gemm_bf16_kernel<true><<<g3, 256, 0, stream>>>(attb, wob, b_out, out, M, D, D);
}
